// TFXLNetRelativeAttention_16106127360556
// MI455X (gfx1250) — matrix operand order repaired and checked by reading, not compiled, not run on hardware
//
#include <hip/hip_runtime.h>
#include <hip/hip_bf16.h>

// ---------------------------------------------------------------------------
// XLNet relative attention, fused, bf16 WMMA + TDM (gfx1250 / MI455X, wave32).
//   D_MODEL=1024, N_HEAD=16, D_HEAD=64, QLEN=1024, MLEN=512, KLEN=1536,
//   RLEN=2560, BSZ=4.  Matrix math via v_wmma_f32_16x16x32_bf16; the V
//   operand of the PV stage is staged into LDS with tensor_load_to_lds.
// ---------------------------------------------------------------------------

#define DM 1024
#define NHD 16
#define DH 64
#define QL 1024
#define ML 512
#define KL 1536
#define RL 2560
#define BS 4
#define BN 64  /* BS*NHD */

typedef __attribute__((ext_vector_type(16))) __bf16 v16bf;
typedef __attribute__((ext_vector_type(8)))  __bf16 v8bf;
typedef __attribute__((ext_vector_type(8)))  float  v8f;
typedef __attribute__((ext_vector_type(4)))  unsigned int v4u;
typedef __attribute__((ext_vector_type(8)))  int v8i;
typedef __attribute__((ext_vector_type(4)))  int v4i;

union U16 { v16bf v; v8bf h[2]; __bf16 e[16]; };

static __device__ __forceinline__ v8f wmma_bf16(v16bf a, v16bf b, v8f c) {
  // D = A(16x32 bf16) x B(32x16 bf16) + C(16x16 f32)
  return __builtin_amdgcn_wmma_f32_16x16x32_bf16(false, a, false, b,
                                                 (short)0, c, false, false);
}

// ---- Tensor Data Mover: 2D tile (bf16) global -> LDS ----------------------
// D# per CDNA5 ISA ch.8: group0 = {count/flags, lds_addr, global_addr, type=2},
// group1 = {data_size=2B, tensor dims, tile dims, dim0 stride}. 2D tile, so
// groups 2/3 are zero. Descriptor operands are wave-uniform -> SGPRs.
static __device__ __forceinline__ void tdm_load_2d_bf16(
    unsigned lds_off, const void* gaddr, unsigned tile_d0, unsigned tile_d1,
    unsigned stride0) {
  unsigned long long ga = (unsigned long long)(size_t)gaddr;
  v4u g0;
  g0[0] = 1u;                                        // count=1 (valid), user
  g0[1] = lds_off;                                   // lds_addr (bytes)
  g0[2] = (unsigned)(ga & 0xffffffffu);              // global_addr[31:0]
  g0[3] = (unsigned)((ga >> 32) & 0x01ffffffu)       // global_addr[56:32]
        | (2u << 30);                                // type = 2 ("image")
  v8i g1;
  g1[0] = (int)(1u << 16);                           // data_size=1 -> 2 bytes
  g1[1] = (int)((tile_d0 & 0xffffu) << 16);          // tensor_dim0[15:0]
  g1[2] = (int)(((tile_d0 >> 16) & 0xffffu)          // tensor_dim0[31:16]
        | ((tile_d1 & 0xffffu) << 16));              // tensor_dim1[15:0]
  g1[3] = (int)(((tile_d1 >> 16) & 0xffffu)          // tensor_dim1[31:16]
        | ((tile_d0 & 0xffffu) << 16));              // tile_dim0
  g1[4] = (int)(tile_d1 & 0xffffu);                  // tile_dim1 (tile_dim2=0)
  g1[5] = (int)stride0;                              // tensor_dim0_stride[31:0]
  g1[6] = 0;                                         // stride0 hi / stride1 lo
  g1[7] = 0;
  v4i z = {};
#if defined(__clang_major__) && (__clang_major__ >= 23)
  v8i z8 = {};
  __builtin_amdgcn_tensor_load_to_lds(g0, g1, z, z, z8, 0);
#else
  __builtin_amdgcn_tensor_load_to_lds(g0, g1, z, z, 0);
#endif
}

// --------------------------- weight conversion -----------------------------
__global__ void k_convT(const float* __restrict__ W, __bf16* __restrict__ Wt) {
  int idx = blockIdx.x * blockDim.x + threadIdx.x;   // 1M elems
  int k = idx >> 10, n = idx & 1023;
  Wt[(size_t)n * DM + k] = (__bf16)W[(size_t)k * DM + n];
}
__global__ void k_conv(const float* __restrict__ W, __bf16* __restrict__ Wb) {
  int idx = blockIdx.x * blockDim.x + threadIdx.x;
  Wb[idx] = (__bf16)W[idx];
}

// --------------------------- projection GEMM -------------------------------
// C[M x 1024] = A(f32, M x 1024) @ Bt^T   (Bt stored [n][k], bf16)
// Output scattered to head layout out[((b*16+nh)*seqLen + s + seqOff)*64 + d]
__global__ __launch_bounds__(256)
void k_proj(const float* __restrict__ A, const __bf16* __restrict__ Bt,
            __bf16* __restrict__ out, int seqLen, int seqOff) {
  const int lane = threadIdx.x & 31;
  const int wave = threadIdx.x >> 5;
  const int ln = lane & 15, kb = lane >> 4;
  const int mbase = blockIdx.x * 128 + wave * 16;
  const int nbase = blockIdx.y * 64;
  const int row = mbase + ln;
  v8f a0 = {}, a1 = {}, a2 = {}, a3 = {};
  for (int k0 = 0; k0 < DM; k0 += 32) {
    U16 af;
    const float* ap = A + (size_t)row * DM + k0 + kb * 16;
#pragma unroll
    for (int p = 0; p < 8; ++p) {
      af.e[p]     = (__bf16)ap[p];        // K = kb*16 + p
      af.e[8 + p] = (__bf16)ap[8 + p];    // K = kb*16 + 8 + p
    }
    const __bf16* bp = Bt + k0 + kb * 16;
    v16bf b0 = *(const v16bf*)(bp + (size_t)(nbase +  0 + ln) * DM);
    v16bf b1 = *(const v16bf*)(bp + (size_t)(nbase + 16 + ln) * DM);
    v16bf b2 = *(const v16bf*)(bp + (size_t)(nbase + 32 + ln) * DM);
    v16bf b3 = *(const v16bf*)(bp + (size_t)(nbase + 48 + ln) * DM);
    a0 = wmma_bf16(af.v, b0, a0);
    a1 = wmma_bf16(af.v, b1, a1);
    a2 = wmma_bf16(af.v, b2, a2);
    a3 = wmma_bf16(af.v, b3, a3);
  }
#pragma unroll
  for (int v = 0; v < 8; ++v) {
    int m = mbase + v + 8 * kb;           // C layout: vgpr v -> rows v, v+8
    int s = m >> 2, b = m & 3;
#pragma unroll
    for (int c = 0; c < 4; ++c) {
      int col = nbase + c * 16 + ln;
      int nh = col >> 6, d = col & 63;
      float val = (c == 0) ? a0[v] : (c == 1) ? a1[v] : (c == 2) ? a2[v] : a3[v];
      out[(((size_t)(b * NHD + nh) * seqLen) + s + seqOff) * DH + d] = (__bf16)val;
    }
  }
}

// --------------------------- small bias GEMVs ------------------------------
// ef1[bn][i][s] = sum_d (q[bn,i,d] + r_s_bias[n,d]) * seg_embed[s,n,d]
__global__ void k_ef1(const __bf16* __restrict__ Q, const float* __restrict__ rsb,
                      const float* __restrict__ seg, float* __restrict__ ef1) {
  int idx = blockIdx.x * blockDim.x + threadIdx.x;  // BN*QL*2
  int s = idx & 1;
  int i = (idx >> 1) & (QL - 1);
  int bn = idx >> 11;
  int nh = bn & 15;
  const __bf16* q = Q + ((size_t)bn * QL + i) * DH;
  float acc = 0.f;
  for (int d = 0; d < DH; ++d)
    acc += ((float)q[d] + rsb[nh * DH + d]) * seg[(s * NHD + nh) * DH + d];
  ef1[idx] = acc;
}

// out[bn][j] = sum_d bias[n,d] * K[bn,j,d]
__global__ void k_biasdot(const float* __restrict__ bias, const __bf16* __restrict__ K,
                          float* __restrict__ out, int len) {
  int idx = blockIdx.x * blockDim.x + threadIdx.x;
  if (idx >= BN * len) return;
  int j = idx % len, bn = idx / len;
  int nh = bn & 15;
  const __bf16* k = K + ((size_t)bn * len + j) * DH;
  float acc = 0.f;
  for (int d = 0; d < DH; ++d) acc += bias[nh * DH + d] * (float)k[d];
  out[idx] = acc;
}

// --------------------------- fused attention -------------------------------
// grid = (64 head-batch pairs, 8 row-blocks of 128); 256 thr = 8 waves;
// each wave owns a 16-row strip and streams j in steps of 32. V tile for
// each step is DMA'd into per-wave LDS by the Tensor Data Mover.
__global__ __launch_bounds__(256)
void k_attn(const __bf16* __restrict__ Q, const __bf16* __restrict__ KH,
            const __bf16* __restrict__ KR, const __bf16* __restrict__ V,
            const float* __restrict__ ef1, const float* __restrict__ rwk,
            const float* __restrict__ rrkr, const float* __restrict__ mask,
            const float* __restrict__ segmat, __bf16* __restrict__ avec) {
  __shared__ __bf16 raw[8][16][48];                 // bd raw band, per wave
  __shared__ __align__(64) __bf16 P[8][16][32];     // probs tile, per wave
  __shared__ float ls[8][16];                       // row sums, per wave
  __shared__ __align__(128) __bf16 Vst[8][32][64];  // TDM-staged V tile
  const int lane = threadIdx.x & 31;
  const int wave = threadIdx.x >> 5;
  const int ln = lane & 15, kb = lane >> 4;
  const int bn = blockIdx.x;
  const int b = bn >> 4, nh = bn & 15;
  const int i0 = blockIdx.y * 128 + wave * 16;
  if (lane < 16) ls[wave][lane] = 0.f;
  const unsigned vst_lds = (unsigned)(unsigned long long)(const void*)&Vst[wave][0][0];

  // Q A-fragments (d = K dim): two 16x32 frags cover d=0..63
  const __bf16* qrow = Q + ((size_t)bn * QL + i0 + ln) * DH;
  U16 qa0, qa1;
  qa0.h[0] = *(const v8bf*)(qrow + kb * 16);
  qa0.h[1] = *(const v8bf*)(qrow + 8 + kb * 16);
  qa1.h[0] = *(const v8bf*)(qrow + 32 + kb * 16);
  qa1.h[1] = *(const v8bf*)(qrow + 40 + kb * 16);

  v8f acc0 = {}, acc1 = {}, acc2 = {}, acc3 = {};
  const __bf16* KHb = KH + (size_t)bn * KL * DH;
  const __bf16* KRb = KR + (size_t)bn * RL * DH;
  const __bf16* Vb  = V  + (size_t)bn * KL * DH;

  for (int j0 = 0; j0 < KL; j0 += 32) {
    // Previous PV reads of Vst must drain before the TDM overwrites it; the
    // compiler cannot see the TDM->LDS dependency, so fence DS explicitly.
    asm volatile("s_wait_dscnt 0x0" ::: "memory");
    // Async DMA: V rows j0..j0+31 (32 x 64 bf16, contiguous) -> Vst[wave].
    tdm_load_2d_bf16(vst_lds, Vb + (size_t)j0 * DH, DH, 32, DH);

    if (j0 + 32 < KL)
      __builtin_prefetch(KHb + (size_t)(j0 + 32 + ln) * DH, 0, 1);

    // ---- ac = Q.K_h^T, two 16-col tiles (overlaps the TDM) ----
    v8f c0 = {}, c1 = {};
    {
      const __bf16* p0 = KHb + (size_t)(j0 + ln) * DH + kb * 16;
      const __bf16* p1 = KHb + (size_t)(j0 + 16 + ln) * DH + kb * 16;
      c0 = wmma_bf16(qa0.v, *(const v16bf*)p0, c0);
      c0 = wmma_bf16(qa1.v, *(const v16bf*)(p0 + 32), c0);
      c1 = wmma_bf16(qa0.v, *(const v16bf*)p1, c1);
      c1 = wmma_bf16(qa1.v, *(const v16bf*)(p1 + 32), c1);
    }

    // ---- bd raw band: 16 rows x 48 cols starting at jr0 ----
    // bd[i,j] = raw[i, (j - j0) + 15 - (i - i0)],  jr0 = j0 + QL - i0 - 15
    const int jr0 = j0 + QL - i0 - 15;
#pragma unroll
    for (int rt = 0; rt < 3; ++rt) {
      int jr = jr0 + rt * 16 + ln;
      if (jr > RL - 1) jr = RL - 1;                 // pad column (unused)
      const __bf16* pr = KRb + (size_t)jr * DH + kb * 16;
      v8f rr = {};
      rr = wmma_bf16(qa0.v, *(const v16bf*)pr, rr);
      rr = wmma_bf16(qa1.v, *(const v16bf*)(pr + 32), rr);
#pragma unroll
      for (int v = 0; v < 8; ++v)
        raw[wave][v + 8 * kb][rt * 16 + ln] = (__bf16)rr[v];
    }

    // ---- scores -> probabilities (no running max; mask = -1e30) ----
#pragma unroll
    for (int t = 0; t < 2; ++t) {
#pragma unroll
      for (int v = 0; v < 8; ++v) {
        int m = v + 8 * kb;
        int jn = t * 16 + ln;
        int i = i0 + m, j = j0 + jn;
        float sc = (t == 0) ? c0[v] : c1[v];
        sc += rwk[(size_t)bn * KL + j];               // r_w_bias . k_h[j]
        sc += (float)raw[wave][m][jn + 15 - m];       // rel-shifted bd
        sc += rrkr[(size_t)bn * RL + (j + QL - i)];   // r_r_bias . k_r[jr]
        size_t ij = (size_t)i * KL + j;
        float e0 = ef1[((size_t)bn * QL + i) * 2];
        float e1 = ef1[((size_t)bn * QL + i) * 2 + 1];
        sc += segmat[(ij * BS + b) * 2] * e0 + segmat[(ij * BS + b) * 2 + 1] * e1;
        sc = sc * 0.125f - 1e30f * mask[ij * BS + b];
        P[wave][m][jn] = (__bf16)__expf(sc);
      }
    }
    // per-row running sum of exp (same wave: DS ops are in order)
    if (lane < 16) {
      float s = 0.f;
#pragma unroll
      for (int jn = 0; jn < 32; ++jn) s += (float)P[wave][lane][jn];
      ls[wave][lane] += s;
    }

    // ---- PV: acc += P(16x32) @ V(32x64), V from TDM-staged LDS ----
    U16 pa;
    pa.h[0] = *(const v8bf*)&P[wave][ln][kb * 16];
    pa.h[1] = *(const v8bf*)&P[wave][ln][kb * 16 + 8];
    __builtin_amdgcn_s_wait_tensorcnt(0);             // V tile landed in LDS
#pragma unroll
    for (int c = 0; c < 4; ++c) {
      v16bf vb;
#pragma unroll
      for (int tt = 0; tt < 16; ++tt)
        vb[tt] = Vst[wave][kb * 16 + tt][c * 16 + ln];
      if (c == 0)      acc0 = wmma_bf16(pa.v, vb, acc0);
      else if (c == 1) acc1 = wmma_bf16(pa.v, vb, acc1);
      else if (c == 2) acc2 = wmma_bf16(pa.v, vb, acc2);
      else             acc3 = wmma_bf16(pa.v, vb, acc3);
    }
  }

  // normalize and scatter to [i][b][nh][d] (row-major M=i*4+b, 1024 cols)
#pragma unroll
  for (int c = 0; c < 4; ++c) {
#pragma unroll
    for (int v = 0; v < 8; ++v) {
      int m = v + 8 * kb;
      int i = i0 + m, d = c * 16 + ln;
      float val = (c == 0) ? acc0[v] : (c == 1) ? acc1[v] : (c == 2) ? acc2[v] : acc3[v];
      val /= ls[wave][m];
      avec[(((size_t)i * BS + b) * NHD + nh) * DH + d] = (__bf16)val;
    }
  }
}

// --------------------------- output GEMM + residual ------------------------
// out[m][h] = sum_c avec[m][c] * Wo[h][c] + resid[m][h]   (Wo bf16 as-is)
__global__ __launch_bounds__(256)
void k_outgemm(const __bf16* __restrict__ A, const __bf16* __restrict__ Bt,
               const float* __restrict__ resid, float* __restrict__ out) {
  const int lane = threadIdx.x & 31;
  const int wave = threadIdx.x >> 5;
  const int ln = lane & 15, kb = lane >> 4;
  const int mbase = blockIdx.x * 128 + wave * 16;
  const int nbase = blockIdx.y * 64;
  const int row = mbase + ln;
  v8f a0 = {}, a1 = {}, a2 = {}, a3 = {};
  for (int k0 = 0; k0 < DM; k0 += 32) {
    U16 af;
    const __bf16* ap = A + (size_t)row * DM + k0 + kb * 16;
    af.h[0] = *(const v8bf*)ap;
    af.h[1] = *(const v8bf*)(ap + 8);
    const __bf16* bp = Bt + k0 + kb * 16;
    v16bf b0 = *(const v16bf*)(bp + (size_t)(nbase +  0 + ln) * DM);
    v16bf b1 = *(const v16bf*)(bp + (size_t)(nbase + 16 + ln) * DM);
    v16bf b2 = *(const v16bf*)(bp + (size_t)(nbase + 32 + ln) * DM);
    v16bf b3 = *(const v16bf*)(bp + (size_t)(nbase + 48 + ln) * DM);
    a0 = wmma_bf16(af.v, b0, a0);
    a1 = wmma_bf16(af.v, b1, a1);
    a2 = wmma_bf16(af.v, b2, a2);
    a3 = wmma_bf16(af.v, b3, a3);
  }
#pragma unroll
  for (int v = 0; v < 8; ++v) {
    int m = mbase + v + 8 * kb;
#pragma unroll
    for (int c = 0; c < 4; ++c) {
      int col = nbase + c * 16 + ln;
      float val = (c == 0) ? a0[v] : (c == 1) ? a1[v] : (c == 2) ? a2[v] : a3[v];
      out[(size_t)m * DM + col] = val + resid[(size_t)m * DM + col];
    }
  }
}

// --------------------------- layernorm -------------------------------------
__global__ __launch_bounds__(256)
void k_ln(const float* __restrict__ x, const float* __restrict__ g,
          const float* __restrict__ bta, float* __restrict__ out) {
  __shared__ float red[256];
  int row = blockIdx.x;
  const float* xr = x + (size_t)row * DM;
  float s = 0.f;
  for (int c = threadIdx.x; c < DM; c += 256) s += xr[c];
  red[threadIdx.x] = s; __syncthreads();
  for (int w = 128; w > 0; w >>= 1) {
    if (threadIdx.x < w) red[threadIdx.x] += red[threadIdx.x + w];
    __syncthreads();
  }
  float mu = red[0] * (1.0f / DM);
  __syncthreads();
  float v = 0.f;
  for (int c = threadIdx.x; c < DM; c += 256) { float d = xr[c] - mu; v += d * d; }
  red[threadIdx.x] = v; __syncthreads();
  for (int w = 128; w > 0; w >>= 1) {
    if (threadIdx.x < w) red[threadIdx.x] += red[threadIdx.x + w];
    __syncthreads();
  }
  float var = red[0] * (1.0f / DM);
  float rs = rsqrtf(var + 1e-12f);
  for (int c = threadIdx.x; c < DM; c += 256)
    out[(size_t)row * DM + c] = (xr[c] - mu) * rs * g[c] + bta[c];
}

// ---------------------------------------------------------------------------
extern "C" void kernel_launch(void* const* d_in, const int* in_sizes, int n_in,
                              void* d_out, int out_size, void* d_ws, size_t ws_size,
                              hipStream_t stream) {
  const float* h    = (const float*)d_in[0];
  const float* r    = (const float*)d_in[1];
  const float* mems = (const float*)d_in[2];
  const float* mask = (const float*)d_in[3];
  const float* seg  = (const float*)d_in[4];
  const float* Wq   = (const float*)d_in[5];
  const float* Wk   = (const float*)d_in[6];
  const float* Wv   = (const float*)d_in[7];
  const float* Wo   = (const float*)d_in[8];
  const float* Wr   = (const float*)d_in[9];
  const float* rwb  = (const float*)d_in[10];
  const float* rrb  = (const float*)d_in[11];
  const float* rsb  = (const float*)d_in[12];
  const float* sege = (const float*)d_in[13];
  const float* gam  = (const float*)d_in[14];
  const float* bet  = (const float*)d_in[15];
  float* outp = (float*)d_out;

  char* w = (char*)d_ws;
  auto alloc = [&](size_t bytes) {
    char* p = w; w += (bytes + 255) & ~(size_t)255; return p;
  };
  __bf16* WtQ  = (__bf16*)alloc((size_t)DM * DM * 2);
  __bf16* WtK  = (__bf16*)alloc((size_t)DM * DM * 2);
  __bf16* WtV  = (__bf16*)alloc((size_t)DM * DM * 2);
  __bf16* WtR  = (__bf16*)alloc((size_t)DM * DM * 2);
  __bf16* WoB  = (__bf16*)alloc((size_t)DM * DM * 2);
  __bf16* qh   = (__bf16*)alloc((size_t)BN * QL * DH * 2);
  __bf16* kh   = (__bf16*)alloc((size_t)BN * KL * DH * 2);
  __bf16* vh   = (__bf16*)alloc((size_t)BN * KL * DH * 2);
  __bf16* kr   = (__bf16*)alloc((size_t)BN * RL * DH * 2);
  __bf16* avec = (__bf16*)alloc((size_t)QL * BS * DM * 2);
  float*  aout = (float*) alloc((size_t)QL * BS * DM * 4);
  float*  ef1  = (float*) alloc((size_t)BN * QL * 2 * 4);
  float*  rwk  = (float*) alloc((size_t)BN * KL * 4);
  float*  rrkr = (float*) alloc((size_t)BN * RL * 4);

  // 1. weight conversion (Wq/Wk/Wv/Wr transposed to [n][k]; Wo as-is)
  k_convT<<<4096, 256, 0, stream>>>(Wq, WtQ);
  k_convT<<<4096, 256, 0, stream>>>(Wk, WtK);
  k_convT<<<4096, 256, 0, stream>>>(Wv, WtV);
  k_convT<<<4096, 256, 0, stream>>>(Wr, WtR);
  k_conv <<<4096, 256, 0, stream>>>(Wo, WoB);

  // 2. projections (cat(mems,h) handled by seq offsets; no concat copy)
  k_proj<<<dim3(32, 16), 256, 0, stream>>>(h,    WtQ, qh, QL, 0);
  k_proj<<<dim3(16, 16), 256, 0, stream>>>(mems, WtK, kh, KL, 0);
  k_proj<<<dim3(32, 16), 256, 0, stream>>>(h,    WtK, kh, KL, ML);
  k_proj<<<dim3(16, 16), 256, 0, stream>>>(mems, WtV, vh, KL, 0);
  k_proj<<<dim3(32, 16), 256, 0, stream>>>(h,    WtV, vh, KL, ML);
  k_proj<<<dim3(80, 16), 256, 0, stream>>>(r,    WtR, kr, RL, 0);

  // 3. folded bias terms
  k_ef1    <<<512, 256, 0, stream>>>(qh, rsb, sege, ef1);
  k_biasdot<<<384, 256, 0, stream>>>(rwb, kh, rwk, KL);
  k_biasdot<<<640, 256, 0, stream>>>(rrb, kr, rrkr, RL);

  // 4. fused attention (ac + rel-shifted bd + ef, softmax, PV via TDM-LDS)
  k_attn<<<dim3(64, 8), 256, 0, stream>>>(qh, kh, kr, vh, ef1, rwk, rrkr,
                                          mask, seg, avec);

  // 5. output projection + residual, then layernorm
  k_outgemm<<<dim3(32, 16), 256, 0, stream>>>(avec, WoB, h, aout);
  k_ln<<<4096, 256, 0, stream>>>(aout, gam, bet, outp);
}
